// BaseRNN_27771258536199
// MI455X (gfx1250) — compile-verified, matrix operand order checked by reading, not run on hardware
//
#include <hip/hip_runtime.h>
#include <hip/hip_bf16.h>
#include <math.h>

// Sizes from the reference
#define B_   256
#define T_   2048
#define OB_  128
#define S_   128
#define H_   128
#define A_   32

typedef __attribute__((ext_vector_type(16))) __bf16 v16bf;
typedef __attribute__((ext_vector_type(8)))  float  v8f;

__device__ __forceinline__ __bf16 f2bf(float f) {
  unsigned u = __builtin_bit_cast(unsigned, f);
  unsigned r = (u + 0x7FFFu + ((u >> 16) & 1u)) >> 16;   // round-to-nearest-even
  unsigned short s = (unsigned short)r;
  return __builtin_bit_cast(__bf16, s);
}

// Hardware transcendental tanh on gfx1250 (V_TANH_F32) when the builtin exists;
// OCML fallback otherwise.
__device__ __forceinline__ float fast_tanh(float x) {
#if __has_builtin(__builtin_amdgcn_tanhf)
  return __builtin_amdgcn_tanhf(x);
#elif __has_builtin(__builtin_amdgcn_tanh_f32)
  return __builtin_amdgcn_tanh_f32(x);
#else
  return tanhf(x);
#endif
}

__device__ __forceinline__ v8f wmma_bf16(v16bf a, v16bf b, v8f c) {
  // D(16x16 f32) = A(16x32 bf16) * B(32x16 bf16) + C
  return __builtin_amdgcn_wmma_f32_16x16x32_bf16(false, a, false, b, (short)0, c,
                                                 false, false);
}

// A-operand 16x32 bf16 tile from an LDS matrix with row stride 128.
// Per ISA 7.12.2 (16-bit A 16x32): lane L -> M = L&15; VGPR r pairs:
//   r<4:  K = 2r + 8*(L>>4);  r>=4: K = 16 + 2(r-4) + 8*(L>>4)
// (contiguous per lane -> compiler emits 2x ds_load_b128)
__device__ __forceinline__ v16bf lds_atile(const __bf16* base, int kb, int nlo, int half) {
  const __bf16* row = base + nlo * 128 + kb * 32;
  v16bf a;
#pragma unroll
  for (int r = 0; r < 8; ++r) {
    int k = ((r < 4) ? 0 : 16) + 2 * (r & 3) + 8 * half;
    a[2 * r]     = row[k];
    a[2 * r + 1] = row[k + 1];
  }
  return a;
}

// B-operand 32x16 tile (cols n0..n0+15) from an fp32 weight matrix, row stride ldn.
// Mirrors the A K-striping with lane -> N. Loaded once (loop-invariant).
__device__ __forceinline__ v16bf glb_btile(const float* W, int ldn, int n0, int kb,
                                           int nlo, int half) {
  int n = n0 + nlo;
  v16bf b;
#pragma unroll
  for (int r = 0; r < 8; ++r) {
    int k = kb * 32 + ((r < 4) ? 0 : 16) + 2 * (r & 3) + 8 * half;
    b[2 * r]     = f2bf(W[(size_t)k * ldn + n]);
    b[2 * r + 1] = f2bf(W[(size_t)(k + 1) * ldn + n]);
  }
  return b;
}

// Waves 8-9 (64 lanes) stream one step's obs tile [16 x 128] fp32 -> LDS bf16.
__device__ __forceinline__ void load_y(const float* __restrict__ obs, int b0, int t,
                                       __bf16 (*dst)[128], int g) {
  int row = g & 15;          // batch row within tile
  int cb  = g >> 4;          // 32-col chunk 0..3
  const float* src = obs + (size_t)(b0 + row) * (T_ * OB_) + (size_t)t * OB_ + cb * 32;
  __bf16* d = &dst[row][cb * 32];
#pragma unroll
  for (int i = 0; i < 8; ++i) {
    float4 v = *(const float4*)(src + 4 * i);
    d[4 * i + 0] = f2bf(v.x);
    d[4 * i + 1] = f2bf(v.y);
    d[4 * i + 2] = f2bf(v.z);
    d[4 * i + 3] = f2bf(v.w);
  }
}

__global__ __launch_bounds__(320, 1)
void rnn_scan_wmma(const float* __restrict__ obs, const float* __restrict__ state0,
                   const float* __restrict__ AK,  const float* __restrict__ BK1,
                   const float* __restrict__ BK2, const float* __restrict__ CK1,
                   const float* __restrict__ DK1, const float* __restrict__ DK2,
                   const float* __restrict__ CK2, const float* __restrict__ DK3,
                   const float* __restrict__ log_stds, float* __restrict__ out) {
  __shared__ __bf16 s_state[2][16][128];   // double-buffered bf16 state
  __shared__ __bf16 s_y[2][16][128];       // double-buffered bf16 obs tile
  __shared__ __bf16 s_z[16][128];          // nonlinearity output

  const int tid  = threadIdx.x;
  const int wave = tid >> 5;
  const int lane = tid & 31;
  const int nlo  = lane & 15;
  const int half = lane >> 4;
  const int b0   = blockIdx.x * 16;        // 16 batch rows per workgroup

  // Loop-invariant weight B-operands held in VGPRs.
  v16bf wB0[4], wB1[4], wB2[4], wB3[4], wB4[4];
  float lstd = 0.0f;
  int n0 = 0;
  if (wave < 8) {                 // state/v waves: own 16 cols of H / S
    n0 = wave * 16;
#pragma unroll
    for (int kb = 0; kb < 4; ++kb) {
      wB0[kb] = glb_btile(CK2, H_, n0, kb, nlo, half);   // phase1: state@CK2
      wB1[kb] = glb_btile(DK3, H_, n0, kb, nlo, half);   // phase1: y@DK3
      wB2[kb] = glb_btile(AK,  S_, n0, kb, nlo, half);   // phase2: state@AK
      wB3[kb] = glb_btile(BK1, S_, n0, kb, nlo, half);   // phase2: z@BK1
      wB4[kb] = glb_btile(BK2, S_, n0, kb, nlo, half);   // phase2: y@BK2
    }
  } else {                        // u waves: own 16 cols of A (=32)
    n0 = (wave - 8) * 16;
#pragma unroll
    for (int kb = 0; kb < 4; ++kb) {
      wB0[kb] = glb_btile(CK1, A_, n0, kb, nlo, half);   // state@CK1
      wB1[kb] = glb_btile(DK1, A_, n0, kb, nlo, half);   // z@DK1
      wB2[kb] = glb_btile(DK2, A_, n0, kb, nlo, half);   // y@DK2
    }
    lstd = log_stds[lane];        // lane 0..31 -> log_std column
  }

  // Prime LDS: state0 (threads 0..255) and y(t=0) (waves 8-9).
  if (tid < 256) {
    int m = tid >> 4, c0 = (tid & 15) * 8;
#pragma unroll
    for (int i = 0; i < 8; ++i)
      s_state[0][m][c0 + i] = f2bf(state0[(size_t)(b0 + m) * S_ + c0 + i]);
  }
  if (wave >= 8) load_y(obs, b0, 0, s_y[0], tid - 256);
  __syncthreads();

  v8f ns = {};                    // final-state carrier (waves 0-7)

  for (int t = 0; t < T_; ++t) {
    const int cur = t & 1, nxt = cur ^ 1;
    const __bf16* st = &s_state[cur][0][0];
    const __bf16* yc = &s_y[cur][0][0];

    if (wave < 8) {
      // ---- phase 1: v = state@CK2 + y@DK3 ; z = 2*tanh(v) - v ----
      v8f accS = {}, accY = {};
#pragma unroll
      for (int kb = 0; kb < 4; ++kb) {
        accS = wmma_bf16(lds_atile(st, kb, nlo, half), wB0[kb], accS);
        accY = wmma_bf16(lds_atile(yc, kb, nlo, half), wB1[kb], accY);
      }
#pragma unroll
      for (int r = 0; r < 8; ++r) {
        float vf = accS[r] + accY[r];
        // L_PHI_INV*(tanh(v) - S_PHI*v) with L_PHI_INV=2, S_PHI=0.5
        float zf = 2.0f * fast_tanh(vf) - vf;
        s_z[r + 8 * half][n0 + nlo] = f2bf(zf);
      }
    } else {
      // ---- overlap: stream next step's obs, prefetch two ahead ----
      if (t + 1 < T_) load_y(obs, b0, t + 1, s_y[nxt], tid - 256);
      if (t + 2 < T_)
        __builtin_prefetch(obs + (size_t)(b0 + (tid & 15)) * (T_ * OB_) +
                               (size_t)(t + 2) * OB_, 0, 1);
    }
    __syncthreads();              // z ready; y(next) ready

    if (wave < 8) {
      // ---- phase 2: new_state = state@AK + z@BK1 + y@BK2 ----
      v8f a0 = {}, a1 = {}, a2 = {};
      const __bf16* zz = &s_z[0][0];
#pragma unroll
      for (int kb = 0; kb < 4; ++kb) {
        a0 = wmma_bf16(lds_atile(st, kb, nlo, half), wB2[kb], a0);
        a1 = wmma_bf16(lds_atile(zz, kb, nlo, half), wB3[kb], a1);
        a2 = wmma_bf16(lds_atile(yc, kb, nlo, half), wB4[kb], a2);
      }
#pragma unroll
      for (int r = 0; r < 8; ++r) {
        float f = a0[r] + a1[r] + a2[r];
        ns[r] = f;
        s_state[nxt][r + 8 * half][n0 + nlo] = f2bf(f);
      }
    } else {
      // ---- u = state@CK1 + z@DK1 + y@DK2 ; write actions + log_stds ----
      v8f a0 = {}, a1 = {}, a2 = {};
      const __bf16* zz = &s_z[0][0];
#pragma unroll
      for (int kb = 0; kb < 4; ++kb) {
        a0 = wmma_bf16(lds_atile(st, kb, nlo, half), wB0[kb], a0);
        a1 = wmma_bf16(lds_atile(zz, kb, nlo, half), wB1[kb], a1);
        a2 = wmma_bf16(lds_atile(yc, kb, nlo, half), wB2[kb], a2);
      }
      const size_t tbase = (size_t)t * 64;
#pragma unroll
      for (int r = 0; r < 8; ++r) {
        int m = r + 8 * half;
        out[(size_t)(b0 + m) * (T_ * 64) + tbase + n0 + nlo] = a0[r] + a1[r] + a2[r];
      }
      const int m0 = (wave - 8) * 8;  // wave8 rows 0-7, wave9 rows 8-15
#pragma unroll
      for (int r = 0; r < 8; ++r)
        out[(size_t)(b0 + m0 + r) * (T_ * 64) + tbase + 32 + lane] = lstd;
    }
    __syncthreads();              // new state committed
  }

  // state_final (fp32, from last phase-2 accumulators), appended after outputs
  if (wave < 8) {
    const size_t ob = (size_t)B_ * T_ * 64;
#pragma unroll
    for (int r = 0; r < 8; ++r)
      out[ob + (size_t)(b0 + r + 8 * half) * S_ + n0 + nlo] = ns[r];
  }
}

extern "C" void kernel_launch(void* const* d_in, const int* in_sizes, int n_in,
                              void* d_out, int out_size, void* d_ws, size_t ws_size,
                              hipStream_t stream) {
  (void)in_sizes; (void)n_in; (void)out_size; (void)d_ws; (void)ws_size;
  const float* obs    = (const float*)d_in[0];
  const float* state0 = (const float*)d_in[1];
  const float* AK     = (const float*)d_in[2];
  const float* BK1    = (const float*)d_in[3];
  const float* BK2    = (const float*)d_in[4];
  const float* CK1    = (const float*)d_in[5];
  const float* DK1    = (const float*)d_in[6];
  const float* DK2    = (const float*)d_in[7];
  const float* CK2    = (const float*)d_in[8];
  const float* DK3    = (const float*)d_in[9];
  const float* lstd   = (const float*)d_in[10];
  float* out = (float*)d_out;

  rnn_scan_wmma<<<dim3(B_ / 16), dim3(320), 0, stream>>>(
      obs, state0, AK, BK1, BK2, CK1, DK1, DK2, CK2, DK3, lstd, out);
}
